// DND_49022756716937
// MI455X (gfx1250) — matrix operand order repaired and checked by reading, not compiled, not run on hardware
//
#include <hip/hip_runtime.h>
#include <hip/hip_bf16.h>

typedef __bf16 bf16_t;
typedef __attribute__((ext_vector_type(16))) __bf16 v16bf;
typedef __attribute__((ext_vector_type(4)))  __bf16 v4bf;
typedef __attribute__((ext_vector_type(8)))  float  v8f;

#define CAP      100000
#define DIMS     128
#define NQ       2048
#define KNN      50
#define NS       16            // key slices
#define TILES    (CAP / 16)    // 6250 N-tiles of 16 keys
#define NC       (NS * KNN)    // 800 candidates per query

// ---------------------------------------------------------------- kernel 1
// Per key row: squared norm + split f32 -> bf16 hi + bf16 lo residual.
__global__ __launch_bounds__(256) void prep_kernel(
    const float* __restrict__ keys, float* __restrict__ knorm,
    bf16_t* __restrict__ khi, bf16_t* __restrict__ klo) {
  int i = blockIdx.x * blockDim.x + threadIdx.x;
  if (i >= CAP) return;
  const float4* r = (const float4*)(keys + (size_t)i * DIMS);
  bf16_t* hrow = khi + (size_t)i * DIMS;
  bf16_t* lrow = klo + (size_t)i * DIMS;
  float s = 0.f;
#pragma unroll
  for (int j = 0; j < DIMS / 4; ++j) {
    float4 v = r[j];
    s += v.x * v.x + v.y * v.y + v.z * v.z + v.w * v.w;
    v4bf h, l;
    float x0 = v.x, x1 = v.y, x2 = v.z, x3 = v.w;
    bf16_t h0 = (bf16_t)x0, h1 = (bf16_t)x1, h2 = (bf16_t)x2, h3 = (bf16_t)x3;
    h[0] = h0; h[1] = h1; h[2] = h2; h[3] = h3;
    l[0] = (bf16_t)(x0 - (float)h0);
    l[1] = (bf16_t)(x1 - (float)h1);
    l[2] = (bf16_t)(x2 - (float)h2);
    l[3] = (bf16_t)(x3 - (float)h3);
    *(v4bf*)(hrow + 4 * j) = h;
    *(v4bf*)(lrow + 4 * j) = l;
  }
  knorm[i] = s;
}

// ---------------------------------------------------------------- kernel 2
// grid: (NQ/64, NS); block: 128 threads = 4 waves; wave w owns queries
// [qt*64 + w*16, +16). All waves walk tiles t == slice (mod NS).
__global__ __launch_bounds__(128) void topk_kernel(
    const float* __restrict__ queries, const bf16_t* __restrict__ khi,
    const bf16_t* __restrict__ klo, const float* __restrict__ knorm,
    float* __restrict__ cand_d, int* __restrict__ cand_i) {
  __shared__ float stage[4][16][17];   // per-wave 16x16 score tile (padded)
  __shared__ float ld[4][16][KNN];     // per-wave sorted top-50 dists
  __shared__ int   li[4][16][KNN];     // per-wave sorted top-50 indices

  const int wave = threadIdx.x >> 5;
  const int lane = threadIdx.x & 31;
  const int l16  = lane & 15;
  const int hf   = lane >> 4;          // 0: lanes 0-15, 1: lanes 16-31
  const int qt   = blockIdx.x;
  const int s    = blockIdx.y;

  // init lists (wave-local, no barrier needed)
  for (int i = lane; i < 16 * KNN; i += 32) {
    ld[wave][i / KNN][i % KNN] = 3.0e38f;
    li[wave][i / KNN][i % KNN] = 0;
  }

  // ---- A fragments (queries) hi/lo bf16, converted once, live in registers.
  // A 16x32 layout used here (contiguous k, not AMD's table order): lane<16 holds M=lane, K 0..15;
  // lane>=16 holds M=lane-16, K 16..31 (matches B).
  const float* qrow = queries + (size_t)(qt * 64 + wave * 16 + l16) * DIMS;
  v16bf a_hi[4], a_lo[4];
#pragma unroll
  for (int kb = 0; kb < 4; ++kb) {
    const float4* r0 = (const float4*)(qrow + kb * 32 + hf * 16);
    const float4* r1 = (const float4*)(qrow + kb * 32 + hf * 16 + 8);
    float x[16];
    float4 v;
    v = r0[0]; x[0] = v.x; x[1] = v.y; x[2] = v.z; x[3] = v.w;
    v = r0[1]; x[4] = v.x; x[5] = v.y; x[6] = v.z; x[7] = v.w;
    v = r1[0]; x[8] = v.x; x[9] = v.y; x[10] = v.z; x[11] = v.w;
    v = r1[1]; x[12] = v.x; x[13] = v.y; x[14] = v.z; x[15] = v.w;
#pragma unroll
    for (int e = 0; e < 16; ++e) {
      bf16_t h = (bf16_t)x[e];
      a_hi[kb][e] = h;
      a_lo[kb][e] = (bf16_t)(x[e] - (float)h);
    }
  }

  // ---- stream N-tiles of this slice; B fragments load pre-split bf16.
  // B 32x16 layout: lane holds N=l16; lanes 0-15 -> K 0..15 of the block,
  // lanes 16-31 -> K 16..31. Keys are row-major == B^T, so each fragment is
  // 16 contiguous bf16 (32B, aligned) per lane.
  for (int t = s; t < TILES; t += NS) {
    const bf16_t* khrow = khi + ((size_t)t * 16 + l16) * DIMS + hf * 16;
    const bf16_t* klrow = klo + ((size_t)t * 16 + l16) * DIMS + hf * 16;
    v8f c = {};
#pragma unroll
    for (int kb = 0; kb < 4; ++kb) {
      v16bf b_hi = *(const v16bf*)(khrow + kb * 32);
      v16bf b_lo = *(const v16bf*)(klrow + kb * 32);
      // fp32-accurate dot via 3 bf16 WMMAs (hi*hi + hi*lo + lo*hi)
      c = __builtin_amdgcn_wmma_f32_16x16x32_bf16(false, a_hi[kb], false, b_hi,
                                                  (short)0, c, false, false);
      c = __builtin_amdgcn_wmma_f32_16x16x32_bf16(false, a_hi[kb], false, b_lo,
                                                  (short)0, c, false, false);
      c = __builtin_amdgcn_wmma_f32_16x16x32_bf16(false, a_lo[kb], false, b_hi,
                                                  (short)0, c, false, false);
    }

    // score = ||k||^2 - 2*q.k  (query norm is constant per row -> irrelevant)
    float kn = knorm[t * 16 + l16];
#pragma unroll
    for (int r = 0; r < 8; ++r)
      stage[wave][hf * 8 + r][l16] = kn - 2.0f * c[r];

    // wave-local top-50 insertion: lanes 0-15 each own one query row
    if (hf == 0) {
      const int m = l16;
      float worst = ld[wave][m][KNN - 1];
#pragma unroll 1
      for (int n = 0; n < 16; ++n) {
        float d = stage[wave][m][n];
        if (d < worst) {
          int pos = KNN - 1;
          while (pos > 0 && ld[wave][m][pos - 1] > d) {
            ld[wave][m][pos] = ld[wave][m][pos - 1];
            li[wave][m][pos] = li[wave][m][pos - 1];
            --pos;
          }
          ld[wave][m][pos] = d;
          li[wave][m][pos] = t * 16 + n;
          worst = ld[wave][m][KNN - 1];
        }
      }
    }
  }

  // flush per-wave lists to the candidate buffer
  for (int i = lane; i < 16 * KNN; i += 32) {
    int m = i / KNN, j = i % KNN;
    int q = qt * 64 + wave * 16 + m;
    size_t off = (size_t)q * NC + (size_t)s * KNN + j;
    cand_d[off] = ld[wave][m][j];
    cand_i[off] = li[wave][m][j];
  }
}

// ---------------------------------------------------------------- kernel 3
__global__ __launch_bounds__(128) void finalize_kernel(
    const float* __restrict__ queries, const float* __restrict__ keys,
    const float* __restrict__ values, const float* __restrict__ cand_d,
    const int* __restrict__ cand_i, float* __restrict__ out) {
  __shared__ float cd[NC];
  __shared__ int   ci[NC];
  __shared__ int   sel[KNN];
  __shared__ float sw[128], swv[128];

  const int q = blockIdx.x;
  const int tid = threadIdx.x;

  for (int i = tid; i < NC; i += blockDim.x) {
    cd[i] = cand_d[(size_t)q * NC + i];
    ci[i] = cand_i[(size_t)q * NC + i];
  }
  __syncthreads();

  // deterministic rank-based selection of the 50 smallest of 800
  for (int j = tid; j < NC; j += blockDim.x) {
    float dj = cd[j];
    int rank = 0;
    for (int k = 0; k < NC; ++k) {
      float dk = cd[k];
      rank += (dk < dj) || (dk == dj && k < j);
    }
    if (rank < KNN) sel[rank] = ci[j];
  }
  __syncthreads();

  // exact f32 recompute of the 50 selected distances (matches reference)
  float w = 0.f, wv = 0.f;
  if (tid < KNN) {
    int idx = sel[tid];
    const float* krow = keys + (size_t)idx * DIMS;
    const float* qrow = queries + (size_t)q * DIMS;
    float sq = 0.f;
#pragma unroll 4
    for (int d = 0; d < DIMS; ++d) {
      float df = qrow[d] - krow[d];
      sq += df * df;
    }
    w = 1.0f / (sq + 1.0e-3f);
    wv = w * values[idx];
  }
  sw[tid] = w;
  swv[tid] = wv;
  __syncthreads();
  if (tid == 0) {
    float W = 0.f, WV = 0.f;
    for (int i = 0; i < KNN; ++i) { W += sw[i]; WV += swv[i]; }
    out[q] = WV / W;
  }
}

// ---------------------------------------------------------------- launcher
extern "C" void kernel_launch(void* const* d_in, const int* in_sizes, int n_in,
                              void* d_out, int out_size, void* d_ws, size_t ws_size,
                              hipStream_t stream) {
  (void)in_sizes; (void)n_in; (void)out_size; (void)ws_size;
  const float* queries = (const float*)d_in[0];
  const float* keys    = (const float*)d_in[1];
  const float* values  = (const float*)d_in[2];
  float* out = (float*)d_out;

  char* ws = (char*)d_ws;
  float*  knorm  = (float*)ws;                                    // 400 KB
  bf16_t* khi    = (bf16_t*)(ws + (1u << 19));                    // 25.6 MB
  bf16_t* klo    = (bf16_t*)(ws + (1u << 19) + (26u << 20));      // 25.6 MB
  float*  cand_d = (float*)(ws + (1u << 19) + (52u << 20));       // 6.55 MB
  int*    cand_i = (int*)(ws + (1u << 19) + (60u << 20));         // 6.55 MB

  prep_kernel<<<(CAP + 255) / 256, 256, 0, stream>>>(keys, knorm, khi, klo);

  dim3 g2(NQ / 64, NS);
  topk_kernel<<<g2, 128, 0, stream>>>(queries, khi, klo, knorm, cand_d, cand_i);

  finalize_kernel<<<NQ, 128, 0, stream>>>(queries, keys, values, cand_d,
                                          cand_i, out);
}
